// NormEMAVectorQuantizer_61065845014873
// MI455X (gfx1250) — compile-verified, matrix operand order checked by reading, not run on hardware
//
#include <hip/hip_runtime.h>
#include <hip/hip_bf16.h>
#include <math.h>

// Problem constants (from reference setup_inputs)
#define D_    256
#define N_    8192
#define BT_   16384      // B*T = 8*2048
#define BETA_ 1.0f

// d_out layout (floats): z_q [0, BT*D), loss [BT*D], indices [BT*D+1, +BT)
#define OUT_ZQ_OFF   0
#define OUT_LOSS_OFF ((size_t)BT_ * D_)
#define OUT_IDX_OFF  ((size_t)BT_ * D_ + 1)

// workspace layout (bytes)
#define WS_ZBF  ((size_t)0)                       // BT*D bf16      = 8388608
#define WS_WBF  ((size_t)8388608)                 // N*D bf16       = 4194304
#define WS_W2   ((size_t)12582912)                // N f32          = 32768
#define WS_IDX  ((size_t)12615680)                // BT i32         = 65536
#define WS_PV   ((size_t)12681216)                // BT*4 f32       = 262144
#define WS_PI   ((size_t)12943360)                // BT*4 i32       = 262144
#define WS_RL   ((size_t)13205504)                // BT f32         = 65536

typedef __attribute__((ext_vector_type(16))) __bf16 v16bf;
typedef __attribute__((ext_vector_type(8)))  float  v8f;

union BfOp { uint4 u[2]; v16bf v; };

__device__ __forceinline__ unsigned short f2bf(float f) {
    unsigned u = __float_as_uint(f);
    u += 0x7FFFu + ((u >> 16) & 1u);          // round-to-nearest-even
    return (unsigned short)(u >> 16);
}

// ---------------------------------------------------------------------------
// Kernel 1: normalize z rows, store bf16. One wave (32 lanes) per row.
__global__ void __launch_bounds__(256) prep_z_kernel(
    const float* __restrict__ z, unsigned short* __restrict__ zbf) {
    const int wave = threadIdx.x >> 5, lane = threadIdx.x & 31;
    const int row  = blockIdx.x * 8 + wave;
    const float* zr = z + (size_t)row * D_;
    float v[8]; float s = 0.f;
#pragma unroll
    for (int j = 0; j < 8; ++j) { v[j] = zr[lane + 32 * j]; s += v[j] * v[j]; }
#pragma unroll
    for (int m = 16; m >= 1; m >>= 1) s += __shfl_xor(s, m, 32);
    const float inv = 1.0f / fmaxf(sqrtf(s), 1e-12f);
    unsigned short* o = zbf + (size_t)row * D_;
#pragma unroll
    for (int j = 0; j < 8; ++j) o[lane + 32 * j] = f2bf(v[j] * inv);
}

// ---------------------------------------------------------------------------
// Kernel 2: codebook -> bf16, plus |w|^2 per code. One wave per row.
__global__ void __launch_bounds__(256) prep_w_kernel(
    const float* __restrict__ w, unsigned short* __restrict__ wbf,
    float* __restrict__ w2) {
    const int wave = threadIdx.x >> 5, lane = threadIdx.x & 31;
    const int row  = blockIdx.x * 8 + wave;
    const float* wr = w + (size_t)row * D_;
    unsigned short* o = wbf + (size_t)row * D_;
    float s = 0.f;
#pragma unroll
    for (int j = 0; j < 8; ++j) {
        float v = wr[lane + 32 * j];
        s += v * v;
        o[lane + 32 * j] = f2bf(v);
    }
#pragma unroll
    for (int m = 16; m >= 1; m >>= 1) s += __shfl_xor(s, m, 32);
    if (lane == 0) w2[row] = s;
}

// ---------------------------------------------------------------------------
// Kernel 3: WMMA distance + running argmin.
// Block: 256 threads = 8 waves; wave owns 16 rows; grid.y splits N 4 ways.
// Double-buffered LDS pipeline: global loads for chunk i+1 overlap the
// 32 WMMAs of chunk i; one barrier per chunk.
__global__ void __launch_bounds__(256) argmin_kernel(
    const unsigned short* __restrict__ zbf, const unsigned short* __restrict__ wbf,
    const float* __restrict__ w2, float* __restrict__ partV, int* __restrict__ partI) {
    __shared__ uint4 smem[2][64 * 34];   // 2 x 64 codewords, 544B pitch

    const int tid  = threadIdx.x;
    const int wave = tid >> 5, lane = tid & 31;
    const int h = lane >> 4, lm = lane & 15;
    const int rowBase = blockIdx.x * 128 + wave * 16;
    const int split   = blockIdx.y;
    const int sr  = tid >> 5;            // staging row handled by this thread
    const int sc  = tid & 31;            // staging col (uint4) within row

    // Resident A tile: 16 rows x 256 (8 chained K-chunks of 32).
    // 16-bit A layout: lane half h holds K {8h..8h+7} then {16+8h..16+8h+7}.
    const uint4* za = (const uint4*)zbf;          // 32 uint4 per row
    const size_t arow = (size_t)(rowBase + lm) * 32;
    BfOp a[8];
#pragma unroll
    for (int c = 0; c < 8; ++c) {
        a[c].u[0] = za[arow + 4 * c + h];
        a[c].u[1] = za[arow + 4 * c + 2 + h];
    }

    float bestv[8]; int besti[8];
#pragma unroll
    for (int r = 0; r < 8; ++r) { bestv[r] = 3.4e38f; besti[r] = 0; }

    const uint4* wb = (const uint4*)wbf;

    // Stage chunk 0 into buffer 0 (each thread: 8 rows, same col).
    {
        const int C0 = split * 32 * 64;
#pragma unroll
        for (int i = 0; i < 8; ++i)
            smem[0][(sr + 8 * i) * 34 + sc] = wb[(size_t)(C0 + sr + 8 * i) * 32 + sc];
    }
    __syncthreads();

    for (int chunk = 0; chunk < 32; ++chunk) {
        const int cur = chunk & 1;
        const int C0  = (split * 32 + chunk) * 64;

        // Issue global loads for the next chunk (waits sink below the compute).
        uint4 stage[8];
        if (chunk + 1 < 32) {
            const int C1 = C0 + 64;
#pragma unroll
            for (int i = 0; i < 8; ++i)
                stage[i] = wb[(size_t)(C1 + sr + 8 * i) * 32 + sc];
        }

        // Compute: two independent accumulation chains, B regs double-buffered.
#pragma unroll
        for (int jp = 0; jp < 2; ++jp) {
            const int j0 = jp * 2;
            const int n0 = C0 + j0 * 16 + lm;
            const int n1 = n0 + 16;
            const float w2n0 = w2[n0];
            const float w2n1 = w2[n1];
            const char* br0 = (const char*)smem[cur] + (size_t)(j0 * 16 + lm) * 544 + 16 * h;
            const char* br1 = br0 + 16 * 544;
            BfOp b0[2], b1[2];
            b0[0].u[0] = *(const uint4*)(br0);
            b0[0].u[1] = *(const uint4*)(br0 + 32);
            b1[0].u[0] = *(const uint4*)(br1);
            b1[0].u[1] = *(const uint4*)(br1 + 32);
            v8f acc0 = {}, acc1 = {};
#pragma unroll
            for (int c = 0; c < 8; ++c) {
                const int nb = (c + 1) & 1;
                if (c < 7) {
                    b0[nb].u[0] = *(const uint4*)(br0 + 64 * (c + 1));
                    b0[nb].u[1] = *(const uint4*)(br0 + 64 * (c + 1) + 32);
                    b1[nb].u[0] = *(const uint4*)(br1 + 64 * (c + 1));
                    b1[nb].u[1] = *(const uint4*)(br1 + 64 * (c + 1) + 32);
                }
                acc0 = __builtin_amdgcn_wmma_f32_16x16x32_bf16(
                        false, a[c].v, false, b0[c & 1].v, (short)0, acc0, false, false);
                acc1 = __builtin_amdgcn_wmma_f32_16x16x32_bf16(
                        false, a[c].v, false, b1[c & 1].v, (short)0, acc1, false, false);
            }
            // D layout: VGPR r -> row (lanes<16: r, lanes>=16: r+8), lane%16 -> col
#pragma unroll
            for (int r = 0; r < 8; ++r) {
                const float s0 = w2n0 - 2.0f * acc0[r];
                if (s0 < bestv[r]) { bestv[r] = s0; besti[r] = n0; }
                const float s1 = w2n1 - 2.0f * acc1[r];
                if (s1 < bestv[r]) { bestv[r] = s1; besti[r] = n1; }
            }
        }

        // Drain the staged loads into the other LDS buffer.
        if (chunk + 1 < 32) {
#pragma unroll
            for (int i = 0; i < 8; ++i)
                smem[cur ^ 1][(sr + 8 * i) * 34 + sc] = stage[i];
        }
        __syncthreads();
    }

    // column reduction: butterfly over the 16 lanes of each half-wave
#pragma unroll
    for (int m = 1; m <= 8; m <<= 1) {
#pragma unroll
        for (int r = 0; r < 8; ++r) {
            const float ov = __shfl_xor(bestv[r], m, 32);
            const int   oi = __shfl_xor(besti[r], m, 32);
            if (ov < bestv[r] || (ov == bestv[r] && oi < besti[r])) {
                bestv[r] = ov; besti[r] = oi;
            }
        }
    }
    if (lm == 0) {
        const int base = rowBase + h * 8;
#pragma unroll
        for (int r = 0; r < 8; ++r) {
            partV[(size_t)(base + r) * 4 + split] = bestv[r];
            partI[(size_t)(base + r) * 4 + split] = besti[r];
        }
    }
}

// ---------------------------------------------------------------------------
// Kernel 4: merge the 4 N-split candidates per row.
__global__ void __launch_bounds__(256) merge_kernel(
    const float* __restrict__ partV, const int* __restrict__ partI,
    int* __restrict__ idxArr) {
    const int row = blockIdx.x * 256 + threadIdx.x;
    float bv = partV[(size_t)row * 4]; int bi = partI[(size_t)row * 4];
#pragma unroll
    for (int s = 1; s < 4; ++s) {
        const float v = partV[(size_t)row * 4 + s];
        const int   i = partI[(size_t)row * 4 + s];
        if (v < bv || (v == bv && i < bi)) { bv = v; bi = i; }
    }
    idxArr[row] = bi;
}

// ---------------------------------------------------------------------------
// Kernel 5: gather z_q (fp32), per-row loss partial, index as float.
__global__ void __launch_bounds__(256) gather_loss_kernel(
    const float* __restrict__ z, const float* __restrict__ w,
    const int* __restrict__ idxArr, float* __restrict__ outZq,
    float* __restrict__ outIdxF, float* __restrict__ rowLoss) {
    const int wave = threadIdx.x >> 5, lane = threadIdx.x & 31;
    const int row  = blockIdx.x * 8 + wave;
    const int idx  = idxArr[row];
    const float* zr = z + (size_t)row * D_;
    float v[8]; float s = 0.f;
#pragma unroll
    for (int j = 0; j < 8; ++j) { v[j] = zr[lane + 32 * j]; s += v[j] * v[j]; }
#pragma unroll
    for (int m = 16; m >= 1; m >>= 1) s += __shfl_xor(s, m, 32);
    const float inv = 1.0f / fmaxf(sqrtf(s), 1e-12f);

    const float* wr = w + (size_t)idx * D_;
    float* o = outZq + (size_t)row * D_;
    float ls = 0.f;
#pragma unroll
    for (int j = 0; j < 8; ++j) {
        const float wv = wr[lane + 32 * j];
        o[lane + 32 * j] = wv;                         // forward value = z_q
        const float d = wv - v[j] * inv;               // z_q - normalized z
        ls += d * d;
    }
#pragma unroll
    for (int m = 16; m >= 1; m >>= 1) ls += __shfl_xor(ls, m, 32);
    if (lane == 0) { rowLoss[row] = ls; outIdxF[row] = (float)idx; }
}

// ---------------------------------------------------------------------------
// Kernel 6: deterministic fixed-order reduction of 16384 row losses.
__global__ void __launch_bounds__(256) finalize_kernel(
    const float* __restrict__ rowLoss, float* __restrict__ outLoss) {
    __shared__ float red[256];
    const int t = threadIdx.x;
    float s = 0.f;
    for (int i = 0; i < 64; ++i) s += rowLoss[t + 256 * i];
    red[t] = s;
    __syncthreads();
    for (int st = 128; st > 0; st >>= 1) {
        if (t < st) red[t] += red[t + st];
        __syncthreads();
    }
    if (t == 0) *outLoss = BETA_ * red[0] / (float)((size_t)BT_ * D_);
}

// ---------------------------------------------------------------------------
extern "C" void kernel_launch(void* const* d_in, const int* in_sizes, int n_in,
                              void* d_out, int out_size, void* d_ws, size_t ws_size,
                              hipStream_t stream) {
    const float* z = (const float*)d_in[0];     // (8,2048,256) f32
    const float* w = (const float*)d_in[1];     // (8192,256) f32 (unit rows)
    float* out = (float*)d_out;

    char* ws = (char*)d_ws;
    unsigned short* zbf = (unsigned short*)(ws + WS_ZBF);
    unsigned short* wbf = (unsigned short*)(ws + WS_WBF);
    float* w2    = (float*)(ws + WS_W2);
    int*   idxA  = (int*)  (ws + WS_IDX);
    float* partV = (float*)(ws + WS_PV);
    int*   partI = (int*)  (ws + WS_PI);
    float* rowL  = (float*)(ws + WS_RL);

    prep_z_kernel<<<BT_ / 8, 256, 0, stream>>>(z, zbf);
    prep_w_kernel<<<N_ / 8, 256, 0, stream>>>(w, wbf, w2);
    argmin_kernel<<<dim3(BT_ / 128, 4), 256, 0, stream>>>(zbf, wbf, w2, partV, partI);
    merge_kernel<<<BT_ / 256, 256, 0, stream>>>(partV, partI, idxA);
    gather_loss_kernel<<<BT_ / 8, 256, 0, stream>>>(
        z, w, idxA, out + OUT_ZQ_OFF, out + OUT_IDX_OFF, rowL);
    finalize_kernel<<<1, 256, 0, stream>>>(rowL, out + OUT_LOSS_OFF);
}
